// LSTMModel_86535001080238
// MI455X (gfx1250) — matrix operand order repaired and checked by reading, not compiled, not run on hardware
//
#include <hip/hip_runtime.h>

typedef __attribute__((ext_vector_type(16))) __bf16 v16bf;
typedef __attribute__((ext_vector_type(8)))  float  v8f;
typedef int v4i_vs __attribute__((__vector_size__(16)));   // <4 x i32>, matches builtin param

#define B_ 128
#define T_ 512
#define I_ 512
#define H_ 512
#define G_ 2048              // 4*H
#define WN (G_ * I_)         // elements per weight matrix (2048*512)

#if defined(__gfx1250__) && __has_builtin(__builtin_amdgcn_global_load_async_to_lds_b128)
#define HAVE_ASYNC_LDS 1
#define GLOBAL_AS __attribute__((address_space(1)))
#define LDS_AS    __attribute__((address_space(3)))
#endif

// ---------------- device helpers ----------------

__device__ __forceinline__ float sigmoidf_(float x) {
  return 1.f / (1.f + __expf(-x));
}
__device__ __forceinline__ float tanh_(float x) {
  return 1.f - 2.f / (__expf(2.f * x) + 1.f);
}

__device__ __forceinline__ void wait_async_lds() {
#if defined(HAVE_ASYNC_LDS)
#if __has_builtin(__builtin_amdgcn_s_wait_asynccnt)
  __builtin_amdgcn_s_wait_asynccnt(0);
#else
  asm volatile("s_wait_asynccnt 0x0" ::: "memory");
#endif
#endif
}

// A fragment (16x32 bf16, MxK) from bf16 LDS (row-major 16 x H_).
// lane L (0..31): row = L&15, off = k0 + (L>>4)*8.
// elements 0..7  <- src[off .. off+7], elements 8..15 <- src[off+16 .. off+23]
__device__ __forceinline__ v16bf a_frag_lds(const __bf16* hb, int row, int off) {
  const __bf16* p = hb + row * H_ + off;
  v16bf a;
#pragma unroll
  for (int e = 0; e < 8; ++e) a[e] = p[e];
#pragma unroll
  for (int e = 0; e < 8; ++e) a[8 + e] = p[16 + e];
  return a;
}

// B fragment (32x16 bf16, KxN) for gates = A @ W^T, W row-major (G_, 512):
// B[k][n] = W[g0+n][k]. lane L: col = L&15, k = k0 + (L>>4)*8 + e (e<8),
// k = k0 + 16 + (L>>4)*8 + (e-8) (e>=8) -> same k order as the A fragment.
__device__ __forceinline__ v16bf b_frag(const __bf16* w, int g0, int k0, int lane) {
  const __bf16* p = w + (size_t)(g0 + (lane & 15)) * I_ + k0 + ((lane >> 4) * 8);
  v16bf b;
#pragma unroll
  for (int e = 0; e < 16; ++e) b[e] = p[e < 8 ? e : 8 + e];
  return b;
}

// ---------------- prep: fp32 weights -> bf16 in ws, fuse biases ----------------

__global__ void prep_weights(const float* __restrict__ wih0, const float* __restrict__ whh0,
                             const float* __restrict__ wih1, const float* __restrict__ whh1,
                             const float* __restrict__ bih0, const float* __restrict__ bhh0,
                             const float* __restrict__ bih1, const float* __restrict__ bhh1,
                             char* __restrict__ ws) {
  __bf16* wb  = (__bf16*)ws;
  float* bias = (float*)(ws + (size_t)4 * WN * 2);
  int i = blockIdx.x * blockDim.x + threadIdx.x;
  if (i < WN) {
    wb[i]          = (__bf16)wih0[i];
    wb[WN + i]     = (__bf16)whh0[i];
    wb[2 * WN + i] = (__bf16)wih1[i];
    wb[3 * WN + i] = (__bf16)whh1[i];
  }
  if (i < G_) {
    bias[i]      = bih0[i] + bhh0[i];
    bias[G_ + i] = bih1[i] + bhh1[i];
  }
}

// ---------------- persistent fused 2-layer LSTM ----------------
// grid = 8 blocks (one per 16 batch rows), block = 1024 threads = 32 wave32.
// Dynamic LDS layout (bytes):
//   gates f32 16x2048 : [0, 131072)
//   c0    f32 16x512  : [131072, 163840)
//   c1    f32 16x512  : [163840, 196608)
//   h0   bf16 16x512  : [196608, 212992)
//   h1   bf16 16x512  : [212992, 229376)
//   x_bf bf16 16x512  : [229376, 245760)
//   xf32  f32 16x512  : [245760, 278528)   (async landing buffer)
//   slen  i32 16      : [278528, 278592)

__global__ void __launch_bounds__(1024)
lstm2_persistent(const float* __restrict__ x, const char* __restrict__ ws,
                 const float* __restrict__ wfc, const float* __restrict__ bfc,
                 float* __restrict__ out) {
  extern __shared__ char smem[];
  float*  gates = (float*)smem;
  float*  c0    = (float*)(smem + 131072);
  float*  c1    = (float*)(smem + 163840);
  __bf16* h0    = (__bf16*)(smem + 196608);
  __bf16* h1    = (__bf16*)(smem + 212992);
  __bf16* x_bf  = (__bf16*)(smem + 229376);
  float*  xf32  = (float*)(smem + 245760);
  int*    slen  = (int*)(smem + 278528);

  const __bf16* Wih0 = (const __bf16*)ws;
  const __bf16* Whh0 = Wih0 + WN;
  const __bf16* Wih1 = Wih0 + 2 * WN;
  const __bf16* Whh1 = Wih0 + 3 * WN;
  const float*  bias0 = (const float*)(ws + (size_t)4 * WN * 2);
  const float*  bias1 = bias0 + G_;

  const int tid  = threadIdx.x;
  const int lane = tid & 31;
  const int wave = tid >> 5;
  const int b0   = blockIdx.x * 16;

  // ---- init state ----
  if (tid < 16) slen[tid] = 0;
  for (int i = tid; i < 16 * H_; i += 1024) {
    c0[i] = 0.f; c1[i] = 0.f;
    h0[i] = (__bf16)0.f; h1[i] = (__bf16)0.f;
  }
  __syncthreads();

  // ---- sequence lengths: count t where sum_i x[b,t,:] != 0 (matches ref) ----
  for (int p = tid; p < 16 * T_; p += 1024) {
    int row = p >> 9, tt = p & (T_ - 1);
    const float* xr = x + ((size_t)(b0 + row) * T_ + tt) * I_;
    float s = 0.f;
    for (int i = 0; i < I_; ++i) s += xr[i];
    if (s != 0.f) atomicAdd(&slen[row], 1);
  }
  __syncthreads();

#if defined(HAVE_ASYNC_LDS)
  // Issue async copy of x_t into xf32: 16 rows x 2KB = 2048 x 16B chunks.
  // Each thread issues 2 chunks; ASYNCcnt tracks completion.
  auto issue_x_async = [&](int t) {
#pragma unroll
    for (int k = 0; k < 2; ++k) {
      int ci    = tid + k * 1024;       // chunk id, 0..2047
      int row   = ci >> 7;              // 128 chunks per row
      int inrow = (ci & 127) * 16;      // byte offset within row
      const char* g = (const char*)(x + ((size_t)(b0 + row) * T_ + t) * I_) + inrow;
      char* l = (char*)xf32 + (size_t)ci * 16;
      __builtin_amdgcn_global_load_async_to_lds_b128(
          (GLOBAL_AS v4i_vs*)g, (LDS_AS v4i_vs*)l, 0, 0);
    }
  };
  issue_x_async(0);
#endif

  const int mrow  = lane & 15;        // A-matrix row owned by this lane
  const int off8  = (lane >> 4) * 8;  // K sub-offset per A layout
  const int nsel  = lane & 15;        // C/D column
  const int mbase = (lane >> 4) * 8;  // C/D row base

#pragma unroll 1
  for (int t = 0; t < T_; ++t) {
    // ---- stage x_t as bf16 in LDS ----
#if defined(HAVE_ASYNC_LDS)
    wait_async_lds();
    __syncthreads();
    for (int i = tid; i < 16 * H_; i += 1024) x_bf[i] = (__bf16)xf32[i];
    __syncthreads();
    if (t + 1 < T_) issue_x_async(t + 1);   // overlap next fetch with compute
#else
    for (int i = tid; i < 16 * H_; i += 1024) {
      int row = i >> 9, col = i & (H_ - 1);
      x_bf[i] = (__bf16)x[((size_t)(b0 + row) * T_ + t) * I_ + col];
    }
    __syncthreads();
#endif

#pragma unroll 1
    for (int layer = 0; layer < 2; ++layer) {
      const __bf16* Wx   = layer ? Wih1 : Wih0;
      const __bf16* Wh   = layer ? Whh1 : Whh0;
      const float*  bias = layer ? bias1 : bias0;
      const __bf16* ain  = layer ? h0 : x_bf;   // input activation
      const __bf16* arec = layer ? h1 : h0;     // recurrent activation

      v8f acc[4];
#pragma unroll
      for (int j = 0; j < 4; ++j) {
        float s = bias[(wave * 4 + j) * 16 + nsel];
#pragma unroll
        for (int r = 0; r < 8; ++r) acc[j][r] = s;
      }

      // ---- input contribution ----
#pragma unroll 1
      for (int ks = 0; ks < I_ / 32; ++ks) {
        v16bf a = a_frag_lds(ain, mrow, ks * 32 + off8);
#pragma unroll
        for (int j = 0; j < 4; ++j) {
          v16bf b = b_frag(Wx, (wave * 4 + j) * 16, ks * 32, lane);
          acc[j] = __builtin_amdgcn_wmma_f32_16x16x32_bf16(
              false, a, false, b, (short)0, acc[j], false, false);
        }
      }
      // ---- recurrent contribution ----
#pragma unroll 1
      for (int ks = 0; ks < H_ / 32; ++ks) {
        v16bf a = a_frag_lds(arec, mrow, ks * 32 + off8);
#pragma unroll
        for (int j = 0; j < 4; ++j) {
          v16bf b = b_frag(Wh, (wave * 4 + j) * 16, ks * 32, lane);
          acc[j] = __builtin_amdgcn_wmma_f32_16x16x32_bf16(
              false, a, false, b, (short)0, acc[j], false, false);
        }
      }

      // ---- spill gate tiles to LDS (C layout: m = mbase+r, n = nsel) ----
#pragma unroll
      for (int j = 0; j < 4; ++j) {
        int g0 = (wave * 4 + j) * 16;
#pragma unroll
        for (int r = 0; r < 8; ++r)
          gates[(mbase + r) * G_ + g0 + nsel] = acc[j][r];
      }
      __syncthreads();

      // ---- elementwise cell update, masked per sequence length ----
      float*  cc = layer ? c1 : c0;
      __bf16* hh = layer ? h1 : h0;
#pragma unroll 1
      for (int q = 0; q < 8; ++q) {
        int u = q * 1024 + tid;              // lanes -> consecutive DWORDs
        int row = u >> 9, hx = u & (H_ - 1);
        float gi = gates[row * G_ + hx];
        float gf = gates[row * G_ + H_ + hx];
        float gg = gates[row * G_ + 2 * H_ + hx];
        float go = gates[row * G_ + 3 * H_ + hx];
        float c  = cc[row * H_ + hx];
        float cn = sigmoidf_(gf) * c + sigmoidf_(gi) * tanh_(gg);
        float hn = sigmoidf_(go) * tanh_(cn);
        bool m = t < slen[row];
        cc[row * H_ + hx] = m ? cn : c;
        __bf16 ho = hh[row * H_ + hx];
        hh[row * H_ + hx] = m ? (__bf16)hn : ho;
      }
      __syncthreads();
    }
  }

  // ---- final FC: out[b, o] = h1[b,:] . wfc[o,:] + bfc[o]  (128x4, trivial) ----
  if (tid < 64) {
    int row = tid >> 2, o = tid & 3;
    float s = bfc[o];
    for (int k = 0; k < H_; ++k)
      s += (float)h1[row * H_ + k] * wfc[o * H_ + k];
    out[(b0 + row) * 4 + o] = s;
  }
}

// ---------------- host entry ----------------

extern "C" void kernel_launch(void* const* d_in, const int* in_sizes, int n_in,
                              void* d_out, int out_size, void* d_ws, size_t ws_size,
                              hipStream_t stream) {
  const float* x    = (const float*)d_in[0];
  const float* wih0 = (const float*)d_in[1];
  const float* whh0 = (const float*)d_in[2];
  const float* bih0 = (const float*)d_in[3];
  const float* bhh0 = (const float*)d_in[4];
  const float* wih1 = (const float*)d_in[5];
  const float* whh1 = (const float*)d_in[6];
  const float* bih1 = (const float*)d_in[7];
  const float* bhh1 = (const float*)d_in[8];
  const float* wfc  = (const float*)d_in[9];
  const float* bfc  = (const float*)d_in[10];
  char*  ws  = (char*)d_ws;
  float* out = (float*)d_out;

  prep_weights<<<(WN + 255) / 256, 256, 0, stream>>>(
      wih0, whh0, wih1, whh1, bih0, bhh0, bih1, bhh1, ws);

  size_t smem = 278592;  // gates + c0/c1 + h0/h1 + x stage + lengths
  lstm2_persistent<<<8, 1024, smem, stream>>>(x, ws, wfc, bfc, out);
}
